// PoseEncoder_81604378624626
// MI455X (gfx1250) — matrix operand order repaired and checked by reading, not compiled, not run on hardware
//
#include <hip/hip_runtime.h>
#include <math.h>

// ---------------------------------------------------------------------------
// CDNA5 (gfx1250) implementation of the PoseEncoder forward pass.
// All GEMMs (conv implicit-GEMM + every LSTM gate GEMM + linear layers)
// lower to v_wmma_f32_16x16x32_f16 (wave32, f32 accumulate).
// Hot loops are guard-free: every GEMM call site has M%64==0, N%32==0, K%32==0.
// ---------------------------------------------------------------------------

typedef _Float16 half16 __attribute__((ext_vector_type(16)));
typedef float    v8f    __attribute__((ext_vector_type(8)));

#define DEVFN __device__ __forceinline__

DEVFN float sigmoid_(float x){ return 1.f/(1.f+__expf(-x)); }
DEVFN float softplus_(float x){ return (x > 20.f) ? x : log1pf(__expf(x)); }

// --- A-matrix (16x32 f16) lane layout: lane holds contiguous K = {kc..kc+15}
//     (kc = k0 + (lane>=16)*16).  Vectorized, unguarded (aligned by contract).
DEVFN void load_a16_fast(const float* __restrict__ A, int lda, int row, int kc,
                         half16& v){
  const float* base = A + (size_t)row*lda + kc;
  float4 x0 = *(const float4*)(base + 0);
  float4 x1 = *(const float4*)(base + 4);
  float4 x2 = *(const float4*)(base + 8);
  float4 x3 = *(const float4*)(base + 12);
  v[0]=(_Float16)x0.x; v[1]=(_Float16)x0.y; v[2]=(_Float16)x0.z; v[3]=(_Float16)x0.w;
  v[4]=(_Float16)x1.x; v[5]=(_Float16)x1.y; v[6]=(_Float16)x1.z; v[7]=(_Float16)x1.w;
  v[8]=(_Float16)x2.x; v[9]=(_Float16)x2.y; v[10]=(_Float16)x2.z; v[11]=(_Float16)x2.w;
  v[12]=(_Float16)x3.x; v[13]=(_Float16)x3.y; v[14]=(_Float16)x3.z; v[15]=(_Float16)x3.w;
}
// --- B-matrix (32x16 f16) lane layout: lane holds contiguous 16-K chunk at
//     kc = k0 + (lane>=16)*16, column N = lane%16 (+ tile base).
DEVFN void load_b16_fast(const float* __restrict__ W, int ldw, int nrow, int kc,
                         half16& v){
  const float* base = W + (size_t)nrow*ldw + kc;
  float4 x0 = *(const float4*)(base + 0);
  float4 x1 = *(const float4*)(base + 4);
  float4 x2 = *(const float4*)(base + 8);
  float4 x3 = *(const float4*)(base + 12);
  v[0]=(_Float16)x0.x; v[1]=(_Float16)x0.y; v[2]=(_Float16)x0.z; v[3]=(_Float16)x0.w;
  v[4]=(_Float16)x1.x; v[5]=(_Float16)x1.y; v[6]=(_Float16)x1.z; v[7]=(_Float16)x1.w;
  v[8]=(_Float16)x2.x; v[9]=(_Float16)x2.y; v[10]=(_Float16)x2.z; v[11]=(_Float16)x2.w;
  v[12]=(_Float16)x3.x; v[13]=(_Float16)x3.y; v[14]=(_Float16)x3.z; v[15]=(_Float16)x3.w;
}

// ---------------------------------------------------------------------------
// Generic GEMM:  D[M,N] = A[M,K] @ W[N,K]^T  (+ A2[M,K2] @ W2[N,K2]^T) + b1 + b2
// 128 threads = 4 waves; each wave computes a 16x32 output tile (2 accumulators
// sharing one A load).  grid = (M/64, N/32).  Requires M%64==0,N%32==0,K%32==0.
// ---------------------------------------------------------------------------
__global__ __launch_bounds__(128)
void wmma_gemm_kernel(const float* __restrict__ A,  int lda,
                      const float* __restrict__ W,  int ldw,
                      const float* __restrict__ A2, int lda2,
                      const float* __restrict__ W2, int ldw2,
                      const float* __restrict__ b1,
                      const float* __restrict__ b2,
                      float* __restrict__ D, int ldd,
                      int N, int K, int K2)
{
  int tid  = threadIdx.x;
  int wid  = tid >> 5;
  int lane = tid & 31;
  int hf   = lane >> 4;        // 0/1
  int l16  = lane & 15;
  int tm = blockIdx.x*4 + wid;
  int n0 = blockIdx.y*32;
  int row   = tm*16 + l16;     // A row (all lanes: same M set, different K chunk)
  int nrow0 = n0 + l16;        // W rows == output columns
  int nrow1 = n0 + 16 + l16;

  v8f acc0 = {}, acc1 = {};
  for (int k0 = 0; k0 < K; k0 += 32){
    half16 a, b0, b1v;
    load_a16_fast(A, lda, row,   k0 + hf*16, a);
    load_b16_fast(W, ldw, nrow0, k0 + hf*16, b0);
    load_b16_fast(W, ldw, nrow1, k0 + hf*16, b1v);
    if (k0 + 32 < K) __builtin_prefetch(&A[(size_t)row*lda + k0 + 32], 0, 1);
    acc0 = __builtin_amdgcn_wmma_f32_16x16x32_f16(false, a, false, b0,
                                                  (short)0, acc0, false, false);
    acc1 = __builtin_amdgcn_wmma_f32_16x16x32_f16(false, a, false, b1v,
                                                  (short)0, acc1, false, false);
  }
  if (A2){
    for (int k0 = 0; k0 < K2; k0 += 32){
      half16 a, b0, b1v;
      load_a16_fast(A2, lda2, row,   k0 + hf*16, a);
      load_b16_fast(W2, ldw2, nrow0, k0 + hf*16, b0);
      load_b16_fast(W2, ldw2, nrow1, k0 + hf*16, b1v);
      acc0 = __builtin_amdgcn_wmma_f32_16x16x32_f16(false, a, false, b0,
                                                    (short)0, acc0, false, false);
      acc1 = __builtin_amdgcn_wmma_f32_16x16x32_f16(false, a, false, b1v,
                                                    (short)0, acc1, false, false);
    }
  }
  int col0 = n0 + l16, col1 = col0 + 16;
  float bv0 = 0.f, bv1 = 0.f;
  if (b1){ bv0 += b1[col0]; bv1 += b1[col1]; }
  if (b2){ bv0 += b2[col0]; bv1 += b2[col1]; }
#pragma unroll
  for (int r = 0; r < 8; ++r){
    int orow = tm*16 + hf*8 + r;     // C/D layout: VGPR r -> M = r + half*8
    D[(size_t)orow*ldd + col0] = acc0[r] + bv0;
    D[(size_t)orow*ldd + col1] = acc1[r] + bv1;
  }
}

// ---------------------------------------------------------------------------
// Implicit-GEMM conv: out[img][co][oy][ox] = post( sum_k patch(m,k)*W[co][k] )
// A(m,k): m -> (img,oy,ox); k = cin*16 + ky*4 + kx  (4x4 kernels, OIHW).
// Branchless bounds (clamped address + select) — no divergent loads.
// Requires Mtot%64==0 and Cout%16==0 (true for all 5 layers).
// ---------------------------------------------------------------------------
DEVFN float conv_fetch(const float* __restrict__ ib, int Cin, int IH, int IW,
                       int iy0, int ix0, int k, int Kd){
  int cin = k >> 4; int kk = k & 15; int ky = kk >> 2, kx = kk & 3;
  int iy = iy0 + ky, ix = ix0 + kx;
  bool ok = (k < Kd) & ((unsigned)iy < (unsigned)IH) & ((unsigned)ix < (unsigned)IW);
  int cinc = min(cin, Cin - 1);
  int iyc  = min(max(iy, 0), IH - 1);
  int ixc  = min(max(ix, 0), IW - 1);
  float t = ib[((size_t)cinc*IH + iyc)*IW + ixc];
  return ok ? t : 0.f;
}

__global__ __launch_bounds__(128)
void conv_wmma_kernel(const float* __restrict__ in, const float* __restrict__ W,
                      const float* __restrict__ g,  const float* __restrict__ bb,
                      float* __restrict__ out,
                      int Cin, int IH, int IW,
                      int Cout, int OH, int OW,
                      int stride, int pad, int K, int relu)
{
  int tid = threadIdx.x;
  int wid = tid >> 5, lane = tid & 31;
  int hf = lane >> 4, l16 = lane & 15;
  int tm = blockIdx.x*4 + wid, tn = blockIdx.y;
  int row = tm*16 + l16;
  int img = row/(OH*OW); int rem = row%(OH*OW); int oy = rem/OW, ox = rem%OW;
  int iy0 = oy*stride - pad, ix0 = ox*stride - pad;
  const float* ibase = in + (size_t)img*Cin*IH*IW;
  int nrow = tn*16 + l16;

  v8f acc = {};
  int Kpad = (K + 31) & ~31;
  for (int k0 = 0; k0 < Kpad; k0 += 32){
    half16 a, b;
    int kca = k0 + hf*16;
#pragma unroll
    for (int e = 0; e < 8; ++e){
      a[e]   = (_Float16)conv_fetch(ibase, Cin, IH, IW, iy0, ix0, kca + e,      K);
      a[e+8] = (_Float16)conv_fetch(ibase, Cin, IH, IW, iy0, ix0, kca + 8 + e,  K);
    }
    int kcb = k0 + hf*16;
#pragma unroll
    for (int e = 0; e < 16; ++e){
      int kk = kcb + e;
      float t = W[(size_t)nrow*K + min(kk, K - 1)];
      b[e] = (_Float16)((kk < K) ? t : 0.f);
    }
    acc = __builtin_amdgcn_wmma_f32_16x16x32_f16(false, a, false, b,
                                                 (short)0, acc, false, false);
  }
  int col = tn*16 + l16;
  float gg = g ? g[col] : 1.f;
  float bv = bb ? bb[col] : 0.f;
#pragma unroll
  for (int r = 0; r < 8; ++r){
    int orow = tm*16 + hf*8 + r;
    int im = orow/(OH*OW); int rem2 = orow%(OH*OW); int oy2 = rem2/OW, ox2 = rem2%OW;
    float v = acc[r]*gg + bv;
    if (relu) v = (v >= 0.f) ? v : 0.2f*v;
    out[(((size_t)im*Cout + col)*OH + oy2)*OW + ox2] = v;
  }
}

// ---------------------------------------------------------------------------
// Pointwise / data-movement kernels
// ---------------------------------------------------------------------------

// LSTM cell: gates[M,4H] (i,f,g,o), updates c,h [M,H]; optional scatter record:
// r = k*Brows + b ; rec[b*strideB + k*strideK + j] = h
__global__ void lstm_cell_kernel(const float* __restrict__ gates,
                                 float* __restrict__ c, float* __restrict__ h,
                                 float* __restrict__ rec, int Brows,
                                 long strideB, long strideK,
                                 int M, int H)
{
  int idx = blockIdx.x*blockDim.x + threadIdx.x;
  if (idx >= M*H) return;
  int r = idx / H, j = idx % H;
  const float* gr = gates + (size_t)r*4*H;
  float gi = gr[j], gf = gr[H + j], gg = gr[2*H + j], go = gr[3*H + j];
  float cn = sigmoid_(gf)*c[idx] + sigmoid_(gi)*tanhf(gg);
  float hn = sigmoid_(go)*tanhf(cn);
  c[idx] = cn; h[idx] = hn;
  if (rec){
    int b = r % Brows, k = r / Brows;
    rec[(size_t)b*strideB + (size_t)k*strideK + j] = hn;
  }
}

// dst[r, :] = [ s1[r,:d1] | s2[r,:d2] | 0-pad ]
__global__ void concat2_kernel(float* __restrict__ dst, int dstride,
                               const float* __restrict__ s1, int s1s, int d1,
                               const float* __restrict__ s2, int s2s, int d2,
                               int rows, int dcols)
{
  int idx = blockIdx.x*blockDim.x + threadIdx.x;
  if (idx >= rows*dcols) return;
  int r = idx / dcols, cidx = idx % dcols;
  float v = 0.f;
  if (cidx < d1) v = s1[(size_t)r*s1s + cidx];
  else if (cidx < d1 + d2) v = s2[(size_t)r*s2s + (cidx - d1)];
  dst[(size_t)r*dstride + cidx] = v;
}

// reprs (B*T, L) -> xf (T, B, L)
__global__ void t_bt_to_tb_kernel(const float* __restrict__ src,
                                  float* __restrict__ dst, int B, int T, int L)
{
  int idx = blockIdx.x*blockDim.x + threadIdx.x;
  if (idx >= B*T*L) return;
  int l = idx % L; int bt = idx / L; int t = bt % T; int b = bt / T;
  dst[((size_t)t*B + b)*L + l] = src[idx];
}

// hidden_in[b,t,k, 0:2H] = [ hs_all[k,t,b,:] | hrs_all[k,T-1-t,b,:] ]
__global__ void build_hidden_in_kernel(const float* __restrict__ hs_all,
                                       const float* __restrict__ hrs_all,
                                       float* __restrict__ dst,
                                       int B, int T, int K, int H)
{
  int idx = blockIdx.x*blockDim.x + threadIdx.x;
  int D = 2*H;
  if (idx >= B*T*K*D) return;
  int cidx = idx % D; int row = idx / D;
  int k = row % K; int bt = row / K; int t = bt % T; int b = bt / T;
  float v;
  if (cidx < H) v = hs_all [(((size_t)k*T + t)        *B + b)*H + cidx];
  else          v = hrs_all[(((size_t)k*T + (T-1-t))  *B + b)*H + (cidx - H)];
  dst[idx] = v;
}

// seq step input (K*B, Dpad): [ sh | m | 0 ];  sh = (t==0)? e : m*e+(1-m)*occ
__global__ void seq_build_kernel(const float* __restrict__ hidden_all,
                                 const float* __restrict__ masks,
                                 const float* __restrict__ occ,
                                 float* __restrict__ dst,
                                 int t, int B, int T, int K, int H, int Dpad)
{
  int idx = blockIdx.x*blockDim.x + threadIdx.x;
  if (idx >= K*B*Dpad) return;
  int cidx = idx % Dpad; int kb = idx / Dpad;
  int k = kb / B, b = kb % B;
  float m = masks[((size_t)b*T + t)*K + k];
  float v = 0.f;
  if (cidx < H){
    float e = hidden_all[(((size_t)b*T + t)*K + k)*H + cidx];
    v = (t == 0) ? e : (m*e + (1.f - m)*occ[(size_t)kb*H + cidx]);
  } else if (cidx == H){
    v = m;
  }
  dst[idx] = v;
}

// pad seq_wih (4H x Din) -> (4H x Dpad), zero fill
__global__ void pad_w_kernel(const float* __restrict__ src, float* __restrict__ dst,
                             int rows, int Din, int Dpad)
{
  int idx = blockIdx.x*blockDim.x + threadIdx.x;
  if (idx >= rows*Dpad) return;
  int c = idx % Dpad, r = idx / Dpad;
  dst[idx] = (c < Din) ? src[(size_t)r*Din + c] : 0.f;
}

// pred init: hp[kb]=hf[b], cp[kb]=cf[b], xx[kb]=enc_out[b,T-1,k,:]
__global__ void pred_init_kernel(float* __restrict__ hp, float* __restrict__ cp,
                                 float* __restrict__ xx,
                                 const float* __restrict__ hfin,
                                 const float* __restrict__ cfin,
                                 const float* __restrict__ enc_out,
                                 int B, int T, int K, int H)
{
  int idx = blockIdx.x*blockDim.x + threadIdx.x;
  if (idx >= K*B*H) return;
  int j = idx % H; int kb = idx / H; int k = kb / B, b = kb % B;
  hp[idx] = hfin[(size_t)b*H + j];
  cp[idx] = cfin[(size_t)b*H + j];
  xx[idx] = enc_out[(((size_t)b*T + (T-1))*K + k)*H + j];
}

// out[m,n] = act( dot(A[m,:K], W[n,:K]) + bias[n] ); one wave per output elem.
__global__ __launch_bounds__(32)
void rowdot_kernel(const float* __restrict__ A, int lda,
                   const float* __restrict__ W, const float* __restrict__ bias,
                   float* __restrict__ out, int M, int N, int Kd, int act)
{
  int o = blockIdx.x;
  if (o >= M*N) return;
  int m = o / N, n = o % N;
  int lane = threadIdx.x & 31;
  float s = 0.f;
  for (int k = lane; k < Kd; k += 32)
    s += A[(size_t)m*lda + k] * W[(size_t)n*Kd + k];
#pragma unroll
  for (int off = 16; off > 0; off >>= 1) s += __shfl_xor(s, off);
  if (lane == 0){
    float v = s + (bias ? bias[n] : 0.f);
    if (act == 1) v = sigmoid_(v);
    else if (act == 2) v = softplus_(v);
    out[(size_t)m*N + n] = v;
  }
}

// ---------------------------------------------------------------------------
// Host orchestration
// ---------------------------------------------------------------------------
static inline dim3 ew_grid(long n){ return dim3((unsigned)((n + 255)/256)); }

static void gemm(hipStream_t s, const float* A, int lda, const float* W, int ldw,
                 const float* A2, int lda2, const float* W2, int ldw2,
                 const float* b1, const float* b2,
                 float* D, int ldd, int M, int N, int K, int K2)
{
  dim3 g((unsigned)(M/64), (unsigned)(N/32));
  wmma_gemm_kernel<<<g, dim3(128), 0, s>>>(A, lda, W, ldw, A2, lda2, W2, ldw2,
                                           b1, b2, D, ldd, N, K, K2);
}

enum {
  IN_VIDEO = 0, IN_SAMPLE,
  CW0, CW1, CW2, CW3, CW4,
  G1, Bc1, G2, Bc2, G3, Bc3,
  ENC_WIH, ENC_WHH, ENC_BIH, ENC_BHH,
  REV_WIH, REV_WHH, REV_BIH, REV_BHH,
  SEQ_WIH, SEQ_WHH, SEQ_BIH, SEQ_BHH,
  PRED_WIH, PRED_WHH, PRED_BIH, PRED_BHH,
  INIT_WIH, INIT_WHH, INIT_BIH, INIT_BHH,
  MASK_WIH, MASK_WHH, MASK_BIH, MASK_BHH,
  ENCL_W, ENCL_B, IMP_W, IMP_B,
  BMU_W, BMU_B, BSIG_W, BSIG_B,
  IMU_W, IMU_B, ISIG_W, ISIG_B,
  MMU_W, MMU_B, MSIG_W, MSIG_B
};

extern "C" void kernel_launch(void* const* d_in, const int* in_sizes, int n_in,
                              void* d_out, int out_size, void* d_ws, size_t ws_size,
                              hipStream_t stream)
{
  (void)in_sizes; (void)n_in; (void)out_size; (void)ws_size;
  const int B = 128, T = 16, L = 128, H = 256, Kc = 3, To = 16, O = 6;
  const int H4 = 4*H;                 // 1024
  const int DinEnc = L + H;           // 384
  const int DseqPad = 288;            // 257 padded to 9*32

  const float* video = (const float*)d_in[IN_VIDEO];
  auto P = [&](int i){ return (const float*)d_in[i]; };

  float* out = (float*)d_out;
  float* out_ibmu  = out;                 // 6144 x 6
  float* out_ibsig = out + 36864;
  float* out_imu   = out + 73728;         // 384 x 6
  float* out_isig  = out + 76032;
  float* out_pbmu  = out + 78336;         // 6144 x 6
  float* out_pbsig = out + 115200;
  float* out_masks = out + 152064;        // 6144

  // ---- workspace bump allocator (floats) --------------------------------
  float* ws = (float*)d_ws;
  size_t off = 0;
  auto alloc = [&](size_t n){ float* p = ws + off; off += n; return p; };

  float* reprs = alloc((size_t)B*T*L);         // 262144
  float* xf    = alloc((size_t)T*B*L);         // 262144
  size_t conv_base = off;
  float* convA = alloc((size_t)2048*16*32*32); // 33.5M
  float* convB = alloc((size_t)2048*32*16*16); // 16.8M
  // post-conv scratch aliases the conv region (conv buffers dead after conv4)
  off = conv_base;
  float* hs_all  = alloc((size_t)Kc*T*B*H);    // (K,T,B,H)
  float* hrs_all = alloc((size_t)Kc*T*B*H);
  float* zerobuf = alloc((size_t)T*B*H);
  float* enc_in  = alloc((size_t)B*DinEnc);
  float* rev_in  = alloc((size_t)B*DinEnc);
  float* gates   = alloc((size_t)2048*H4);     // shared gates scratch (max M=2048)
  float* hf = alloc((size_t)B*H); float* cf = alloc((size_t)B*H);
  float* hr = alloc((size_t)B*H); float* cr = alloc((size_t)B*H);
  float* hidden_in  = alloc((size_t)B*T*Kc*2*H);
  float* hidden_all = alloc((size_t)B*T*Kc*H);
  float* hm = alloc((size_t)B*T*H); float* cm = alloc((size_t)B*T*H);
  float* mr = alloc((size_t)B*T*Kc*H);
  float* hsq = alloc((size_t)Kc*B*H); float* csq = alloc((size_t)Kc*B*H);
  float* occ = alloc((size_t)Kc*B*H);
  float* seq_in = alloc((size_t)Kc*B*DseqPad);
  float* wpad   = alloc((size_t)H4*DseqPad);
  float* enc_out = alloc((size_t)B*T*Kc*H);    // (B,T,K,H)
  float* irbuf   = alloc((size_t)B*Kc*H);      // (B,K,H)
  float* hi = alloc((size_t)B*H); float* ci = alloc((size_t)B*H);
  float* hp = alloc((size_t)Kc*B*H); float* cp = alloc((size_t)Kc*B*H);
  float* xx = alloc((size_t)Kc*B*H);
  float* predout = alloc((size_t)B*To*Kc*H);   // (B,To,K,H)

  const size_t stateBytes = (size_t)B*H*sizeof(float);

  // ---- conv encoder (implicit-GEMM WMMA) --------------------------------
  {
    struct CC { const float* in; const float* w; const float* g; const float* b;
                float* o; int Ci, IH, IW, Co, OH, OW, st, pd, K, relu; };
    CC cc[5] = {
      { video, P(CW0), nullptr, nullptr, convA,   1, 64, 64,  16, 32, 32, 2, 1,   16, 1 },
      { convA, P(CW1), P(G1),   P(Bc1),  convB,  16, 32, 32,  32, 16, 16, 2, 1,  256, 1 },
      { convB, P(CW2), P(G2),   P(Bc2),  convA,  32, 16, 16,  64,  8,  8, 2, 1,  512, 1 },
      { convA, P(CW3), P(G3),   P(Bc3),  convB,  64,  8,  8, 128,  4,  4, 2, 1, 1024, 1 },
      { convB, P(CW4), nullptr, nullptr, reprs, 128,  4,  4, 128,  1,  1, 1, 0, 2048, 0 },
    };
    for (int i = 0; i < 5; ++i){
      int Mtot = 2048*cc[i].OH*cc[i].OW;
      dim3 g((unsigned)(Mtot/64), (unsigned)(cc[i].Co/16));
      conv_wmma_kernel<<<g, dim3(128), 0, stream>>>(cc[i].in, cc[i].w, cc[i].g, cc[i].b,
          cc[i].o, cc[i].Ci, cc[i].IH, cc[i].IW, cc[i].Co, cc[i].OH, cc[i].OW,
          cc[i].st, cc[i].pd, cc[i].K, cc[i].relu);
    }
  }
  t_bt_to_tb_kernel<<<ew_grid((long)B*T*L), 256, 0, stream>>>(reprs, xf, B, T, L);

  // ---- bidirectional component scan (K components x T steps) ------------
  hipMemsetAsync(zerobuf, 0, (size_t)T*B*H*sizeof(float), stream);
  for (int k = 0; k < Kc; ++k){
    const float* prevH  = (k == 0) ? zerobuf : hs_all  + (size_t)(k-1)*T*B*H;
    const float* prevHr = (k == 0) ? zerobuf : hrs_all + (size_t)(k-1)*T*B*H;
    hipMemsetAsync(hf, 0, stateBytes, stream);
    hipMemsetAsync(cf, 0, stateBytes, stream);
    for (int j = 0; j < T; ++j){
      // forward LSTM step
      concat2_kernel<<<ew_grid((long)B*DinEnc), 256, 0, stream>>>(
          enc_in, DinEnc, xf + (size_t)j*B*L, L, L, prevH + (size_t)j*B*H, H, H, B, DinEnc);
      gemm(stream, enc_in, DinEnc, P(ENC_WIH), DinEnc, hf, H, P(ENC_WHH), H,
           P(ENC_BIH), P(ENC_BHH), gates, H4, B, H4, DinEnc, H);
      lstm_cell_kernel<<<ew_grid((long)B*H), 256, 0, stream>>>(
          gates, cf, hf, hs_all + ((size_t)k*T + j)*B*H, B, H, 0, B, H);
      if (j == 0){
        hipMemcpyAsync(hr, hf, stateBytes, hipMemcpyDeviceToDevice, stream);
        hipMemcpyAsync(cr, cf, stateBytes, hipMemcpyDeviceToDevice, stream);
      }
      // reverse LSTM step (x_r[j] = xf[T-1-j])
      concat2_kernel<<<ew_grid((long)B*DinEnc), 256, 0, stream>>>(
          rev_in, DinEnc, xf + (size_t)(T-1-j)*B*L, L, L, prevHr + (size_t)j*B*H, H, H, B, DinEnc);
      gemm(stream, rev_in, DinEnc, P(REV_WIH), DinEnc, hr, H, P(REV_WHH), H,
           P(REV_BIH), P(REV_BHH), gates, H4, B, H4, DinEnc, H);
      lstm_cell_kernel<<<ew_grid((long)B*H), 256, 0, stream>>>(
          gates, cr, hr, hrs_all + ((size_t)k*T + j)*B*H, B, H, 0, B, H);
    }
  } // hf/cf now hold h_fin/c_fin

  // ---- hidden_all = concat(hsa, rr) @ encl_w^T + encl_b ------------------
  build_hidden_in_kernel<<<ew_grid((long)B*T*Kc*2*H), 256, 0, stream>>>(
      hs_all, hrs_all, hidden_in, B, T, Kc, H);
  gemm(stream, hidden_in, 2*H, P(ENCL_W), 2*H, nullptr, 0, nullptr, 0,
       P(ENCL_B), nullptr, hidden_all, H, B*T*Kc, H, 2*H, 0);

  // ---- mask LSTM over K (batch B*T) + masks head -------------------------
  hipMemsetAsync(hm, 0, (size_t)B*T*H*sizeof(float), stream);
  hipMemsetAsync(cm, 0, (size_t)B*T*H*sizeof(float), stream);
  for (int k = 0; k < Kc; ++k){
    gemm(stream, hidden_all + (size_t)k*H, Kc*H, P(MASK_WIH), H, hm, H, P(MASK_WHH), H,
         P(MASK_BIH), P(MASK_BHH), gates, H4, B*T, H4, H, H);
    lstm_cell_kernel<<<ew_grid((long)B*T*H), 256, 0, stream>>>(
        gates, cm, hm, mr + (size_t)k*H, B*T, Kc*H, 0, B*T, H);
  }
  rowdot_kernel<<<dim3(B*T*Kc), dim3(32), 0, stream>>>(
      mr, H, P(MMU_W), P(MMU_B), out_masks, B*T*Kc, 1, H, /*sigmoid*/1);

  // ---- padded seq weights -------------------------------------------------
  pad_w_kernel<<<ew_grid((long)H4*DseqPad), 256, 0, stream>>>(
      P(SEQ_WIH), wpad, H4, H + 1, DseqPad);

  // ---- seq_comp: scan over T, batched over (K*B) --------------------------
  hipMemsetAsync(hsq, 0, (size_t)Kc*B*H*sizeof(float), stream);
  hipMemsetAsync(csq, 0, (size_t)Kc*B*H*sizeof(float), stream);
  for (int t = 0; t < T; ++t){
    gemm(stream, hsq, H, P(IMP_W), H, nullptr, 0, nullptr, 0,
         P(IMP_B), nullptr, occ, H, Kc*B, H, H, 0);
    seq_build_kernel<<<ew_grid((long)Kc*B*DseqPad), 256, 0, stream>>>(
        hidden_all, out_masks, occ, seq_in, t, B, T, Kc, H, DseqPad);
    gemm(stream, seq_in, DseqPad, wpad, DseqPad, hsq, H, P(SEQ_WHH), H,
         P(SEQ_BIH), P(SEQ_BHH), gates, H4, Kc*B, H4, DseqPad, H);
    lstm_cell_kernel<<<ew_grid((long)Kc*B*H), 256, 0, stream>>>(
        gates, csq, hsq, enc_out + (size_t)t*Kc*H, B, (long)T*Kc*H, H, Kc*B, H);
  }

  // ---- init LSTM over K (batch B) ----------------------------------------
  hipMemsetAsync(hi, 0, stateBytes, stream);
  hipMemsetAsync(ci, 0, stateBytes, stream);
  for (int k = 0; k < Kc; ++k){
    // x_k[b,:] = enc_out[b, 0, k, :]  (row stride T*K*H)
    gemm(stream, enc_out + (size_t)k*H, T*Kc*H, P(INIT_WIH), H, hi, H, P(INIT_WHH), H,
         P(INIT_BIH), P(INIT_BHH), gates, H4, B, H4, H, H);
    lstm_cell_kernel<<<ew_grid((long)B*H), 256, 0, stream>>>(
        gates, ci, hi, irbuf + (size_t)k*H, B, (long)Kc*H, 0, B, H);
  }

  // ---- prediction scan: To steps, batched over (K*B) ----------------------
  pred_init_kernel<<<ew_grid((long)Kc*B*H), 256, 0, stream>>>(
      hp, cp, xx, hf, cf, enc_out, B, T, Kc, H);
  for (int t = 0; t < To; ++t){
    const float* xin = (t == 0) ? xx : hp;   // carry: xx becomes previous h
    gemm(stream, xin, H, P(PRED_WIH), H, hp, H, P(PRED_WHH), H,
         P(PRED_BIH), P(PRED_BHH), gates, H4, Kc*B, H4, H, H);
    lstm_cell_kernel<<<ew_grid((long)Kc*B*H), 256, 0, stream>>>(
        gates, cp, hp, predout + (size_t)t*Kc*H, B, (long)To*Kc*H, H, Kc*B, H);
  }

  // ---- output heads (small N: wave-reduction rowdots) ---------------------
  rowdot_kernel<<<dim3(B*T*Kc*O), dim3(32), 0, stream>>>(
      enc_out, H, P(BMU_W),  P(BMU_B),  out_ibmu,  B*T*Kc, O, H, 0);
  rowdot_kernel<<<dim3(B*T*Kc*O), dim3(32), 0, stream>>>(
      enc_out, H, P(BSIG_W), P(BSIG_B), out_ibsig, B*T*Kc, O, H, 2);
  rowdot_kernel<<<dim3(B*Kc*O), dim3(32), 0, stream>>>(
      irbuf,   H, P(IMU_W),  P(IMU_B),  out_imu,   B*Kc,   O, H, 0);
  rowdot_kernel<<<dim3(B*Kc*O), dim3(32), 0, stream>>>(
      irbuf,   H, P(ISIG_W), P(ISIG_B), out_isig,  B*Kc,   O, H, 2);
  rowdot_kernel<<<dim3(B*To*Kc*O), dim3(32), 0, stream>>>(
      predout, H, P(BMU_W),  P(BMU_B),  out_pbmu,  B*To*Kc, O, H, 0);
  rowdot_kernel<<<dim3(B*To*Kc*O), dim3(32), 0, stream>>>(
      predout, H, P(BSIG_W), P(BSIG_B), out_pbsig, B*To*Kc, O, H, 2);
}
